// CrossAttentionPool_6055903888143
// MI455X (gfx1250) — compile-verified, matrix operand order checked by reading, not run on hardware
//
#include <hip/hip_runtime.h>
#include <cstdint>

#define B_ 32
#define R_ 8192
#define D_ 512
#define H_ 8
#define ROWS 128
#define TILES (R_/ROWS)   /* 64 */
#define KSTR 520          /* padded bf16 row stride for K tile */
#define ESTR 132          /* padded f32 row stride for exp-scores */

typedef __attribute__((ext_vector_type(16))) __bf16 v16bf;
typedef __attribute__((ext_vector_type(8)))  float  v8f;

union BF16x16 { v16bf v; struct { uint4 lo; uint4 hi; } q; };

__device__ __forceinline__ unsigned short f2bf(float x) {
  unsigned int u = __float_as_uint(x);
  u += 0x7FFFu + ((u >> 16) & 1u);           // round-to-nearest-even
  return (unsigned short)(u >> 16);
}
__device__ __forceinline__ float bf2f(unsigned short h) {
  return __uint_as_float(((unsigned int)h) << 16);
}

// ---------------------------------------------------------------------------
// K1: q = r@Wq^T + bq; fold q and 1/sqrt(dk) into per-head score matrix
//     A[b,h,d] = (sum_j q[b,h,j]*Wk[h*64+j,d])/8  (bf16, heads 8..15 zero-pad)
//     c[b,h]   = (sum_j q[b,h,j]*bk[h*64+j])/8
// ---------------------------------------------------------------------------
__global__ __launch_bounds__(256) void qcombine_kernel(
    const float* __restrict__ r, const float* __restrict__ Wq, const float* __restrict__ bq,
    const float* __restrict__ Wk, const float* __restrict__ bk,
    unsigned short* __restrict__ Aws, float* __restrict__ cpr) {
  __shared__ float rb[D_];
  __shared__ float qv[D_];
  const int b = blockIdx.x, t = threadIdx.x;
  for (int i = t; i < D_; i += 256) rb[i] = r[b*D_ + i];
  __syncthreads();
  for (int i = t; i < D_; i += 256) {
    const float* w = Wq + (size_t)i*D_;
    float a = bq[i];
    for (int d = 0; d < D_; ++d) a = fmaf(rb[d], w[d], a);
    qv[i] = a;
  }
  __syncthreads();
  for (int idx = t; idx < 16*D_; idx += 256) {
    const int h = idx >> 9, d = idx & (D_-1);
    float v = 0.f;
    if (h < H_) {
      const float* wk = Wk + (size_t)(h*64)*D_ + d;
      for (int j = 0; j < 64; ++j) v = fmaf(qv[h*64+j], wk[(size_t)j*D_], v);
      v *= 0.125f;
    }
    Aws[(size_t)(b*16 + h)*D_ + d] = f2bf(v);
  }
  if (t < H_) {
    float c = 0.f;
    for (int j = 0; j < 64; ++j) c = fmaf(qv[t*64+j], bk[t*64+j], c);
    cpr[b*H_ + t] = c * 0.125f;
  }
}

// ---------------------------------------------------------------------------
// K2 (the single pass over K, 537 MB):
//  stage 128x512 K-tile as bf16 in LDS;
//  pass 1: scores = Ktile @ A[b]^T via v_wmma_f32_16x16x32_bf16 (8 waves x 16 wmma)
//          -> exp (with +c, mask) into LDS and unnormalized attn output
//  pass 2: tile-partial pooled sums tpart[h][d] = sum_r exp*Ktile (from resident LDS)
// ---------------------------------------------------------------------------
__global__ __launch_bounds__(256) void score_pool_kernel(
    const float* __restrict__ Kg, const unsigned char* __restrict__ mask,
    const unsigned short* __restrict__ Aws, const float* __restrict__ cpr,
    float* __restrict__ attn, float* __restrict__ Tpart) {
  extern __shared__ char smem[];
  unsigned short* Kt  = (unsigned short*)smem;               // 128*520 bf16 = 133120 B
  unsigned short* Abf = (unsigned short*)(smem + 133120);    // 16*512 bf16 = 16384 B
  float* esl          = (float*)(smem + 149504);             // 8*132 f32  = 4224 B
  float* cpl          = (float*)(smem + 153728);             // 8 f32

  const int tid  = threadIdx.x;
  const int b    = blockIdx.y;
  const int tile = blockIdx.x;
  const int r0   = tile * ROWS;
  const float* Kb = Kg + ((size_t)b*R_ + r0)*D_;

  // stage K tile (coalesced float4 loads, bf16 convert)
  for (int f = tid; f < ROWS*(D_/4); f += 256) {
    const int row = f >> 7;
    const int c4  = f & 127;
    const float4 v = *(const float4*)(Kb + (size_t)row*D_ + c4*4);
    uint2 p;
    p.x = ((unsigned)f2bf(v.y) << 16) | (unsigned)f2bf(v.x);
    p.y = ((unsigned)f2bf(v.w) << 16) | (unsigned)f2bf(v.z);
    *(uint2*)(Kt + (size_t)row*KSTR + c4*4) = p;
  }
  {
    const uint4* src = (const uint4*)(Aws + (size_t)b*16*D_);
    uint4* dst = (uint4*)Abf;
    for (int i = tid; i < (16*D_)/8; i += 256) dst[i] = src[i];
  }
  if (tid < H_) cpl[tid] = cpr[b*H_ + tid];
  __syncthreads();

  // ---- pass 1: WMMA scores ----
  const int lane = tid & 31;
  const int ln   = lane & 15;          // M (A) / N (B) index
  const int hf   = lane >> 4;          // half-wave selects K sub-chunk
  const int m0   = (tid >> 5) << 4;    // 16-row tile per wave
  v8f acc;
  #pragma unroll
  for (int i = 0; i < 8; ++i) acc[i] = 0.f;
  const unsigned short* Krow = Kt  + (size_t)(m0 + ln)*KSTR;
  const unsigned short* Arow = Abf + (size_t)ln*D_;
  for (int kb = 0; kb < D_; kb += 32) {
    BF16x16 a, bm;
    a.q.lo  = *(const uint4*)(Krow + kb + hf*8);       // K=hf*8..+7
    a.q.hi  = *(const uint4*)(Krow + kb + 16 + hf*8);  // K=16+hf*8..+7
    bm.q.lo = *(const uint4*)(Arow + kb + hf*8);
    bm.q.hi = *(const uint4*)(Arow + kb + 16 + hf*8);
    acc = __builtin_amdgcn_wmma_f32_16x16x32_bf16(false, a.v, false, bm.v,
                                                  (short)0, acc, false, false);
  }
  if (ln < H_) {                        // valid heads only (N = ln)
    const float cp = cpl[ln];
    #pragma unroll
    for (int j = 0; j < 8; ++j) {       // VGPR j -> M = j + 8*hf
      const int rl = m0 + j + 8*hf;
      const int rg = r0 + rl;
      const float e = mask[(size_t)b*R_ + rg] ? __expf(acc[j] + cp) : 0.f;
      esl[ln*ESTR + rl] = e;
      attn[((size_t)(b*H_ + ln))*R_ + rg] = e;   // unnormalized; K4 divides
    }
  }
  __syncthreads();

  // ---- pass 2: per-tile pooled partials from resident LDS tile ----
  const int d0 = tid * 2;               // each thread owns 2 d-columns, all 8 heads
  float av[16];
  #pragma unroll
  for (int i = 0; i < 16; ++i) av[i] = 0.f;
  for (int rl = 0; rl < ROWS; ++rl) {
    const float k0 = bf2f(Kt[(size_t)rl*KSTR + d0]);
    const float k1 = bf2f(Kt[(size_t)rl*KSTR + d0 + 1]);
    #pragma unroll
    for (int h = 0; h < H_; ++h) {
      const float e = esl[h*ESTR + rl];
      av[2*h]   = fmaf(e, k0, av[2*h]);
      av[2*h+1] = fmaf(e, k1, av[2*h+1]);
    }
  }
  float* tp = Tpart + (size_t)(b*TILES + tile)*H_*D_;
  #pragma unroll
  for (int h = 0; h < H_; ++h) {
    tp[(size_t)h*D_ + d0]     = av[2*h];
    tp[(size_t)h*D_ + d0 + 1] = av[2*h+1];
  }
}

// K3: deterministic fixed-order reduction of per-tile pooled partials
__global__ __launch_bounds__(256) void treduce_kernel(
    const float* __restrict__ Tpart, float* __restrict__ Tws) {
  const int idx = blockIdx.x*256 + threadIdx.x;       // < B*H*D = 131072
  const int b  = idx >> 12;
  const int hd = idx & 4095;
  const float* p = Tpart + (size_t)b*TILES*H_*D_ + hd;
  float s = 0.f;
  for (int t = 0; t < TILES; ++t) s += p[(size_t)t*H_*D_];
  Tws[idx] = s;
}

// K4: per-(b,h) softmax denominator (tree reduce) + in-place attn normalize
__global__ __launch_bounds__(256) void softmax_norm_kernel(
    float* __restrict__ attn, float* __restrict__ Ssum) {
  __shared__ float red[256];
  const int bh = blockIdx.x, t = threadIdx.x;
  float* a = attn + (size_t)bh*R_;
  float s = 0.f;
  for (int i = t; i < R_; i += 256) s += a[i];
  red[t] = s; __syncthreads();
  for (int k = 128; k > 0; k >>= 1) { if (t < k) red[t] += red[t+k]; __syncthreads(); }
  const float inv = 1.f / red[0];
  if (t == 0) Ssum[bh] = red[0];
  for (int i = t; i < R_; i += 256) a[i] *= inv;
}

// K5: pooled = ((t/S)@Wv^T + bv) @ Wo^T + bo   (tiny: 32x512)
__global__ __launch_bounds__(256) void final_kernel(
    const float* __restrict__ Tws, const float* __restrict__ Ssum,
    const float* __restrict__ Wv, const float* __restrict__ bv,
    const float* __restrict__ Wo, const float* __restrict__ bo,
    float* __restrict__ out) {
  __shared__ float u[D_];
  const int b = blockIdx.x, t = threadIdx.x;
  for (int i = t; i < D_; i += 256) {
    const int h = i >> 6;
    const float inv = 1.f / Ssum[b*H_ + h];
    const float* tv = Tws + (size_t)(b*H_ + h)*D_;
    const float* w  = Wv + (size_t)i*D_;
    float a = 0.f;
    for (int d = 0; d < D_; ++d) a = fmaf(tv[d], w[d], a);
    u[i] = a * inv + bv[i];
  }
  __syncthreads();
  for (int i = t; i < D_; i += 256) {
    const float* w = Wo + (size_t)i*D_;
    float a = bo[i];
    for (int d = 0; d < D_; ++d) a = fmaf(u[d], w[d], a);
    out[(size_t)b*D_ + i] = a;
  }
}

extern "C" void kernel_launch(void* const* d_in, const int* in_sizes, int n_in,
                              void* d_out, int out_size, void* d_ws, size_t ws_size,
                              hipStream_t stream) {
  (void)in_sizes; (void)n_in; (void)out_size; (void)ws_size;
  const float* r  = (const float*)d_in[0];
  const float* Kg = (const float*)d_in[1];
  const unsigned char* mask = (const unsigned char*)d_in[2];
  const float* Wq = (const float*)d_in[3];
  const float* bq = (const float*)d_in[4];
  const float* Wk = (const float*)d_in[5];
  const float* bk = (const float*)d_in[6];
  const float* Wv = (const float*)d_in[7];
  const float* bv = (const float*)d_in[8];
  const float* Wo = (const float*)d_in[9];
  const float* bo = (const float*)d_in[10];

  float* out_pooled = (float*)d_out;             // [32,512]
  float* out_attn   = out_pooled + B_*D_;        // [32,8,8192]

  char* ws = (char*)d_ws;
  unsigned short* Aws = (unsigned short*)(ws);               // 524288 B
  float* cpr   = (float*)(ws + 524288);                      // 1024 B
  float* Ssum  = (float*)(ws + 525312);                      // 1024 B
  float* Tws   = (float*)(ws + 526336);                      // 524288 B
  float* Tpart = (float*)(ws + 1050624);                     // 33.5 MB

  const size_t SMEM = 153760;  // Ktile(bf16) + A(bf16) + exp-scores + c
  hipFuncSetAttribute(reinterpret_cast<const void*>(score_pool_kernel),
                      hipFuncAttributeMaxDynamicSharedMemorySize, (int)SMEM);

  qcombine_kernel<<<B_, 256, 0, stream>>>(r, Wq, bq, Wk, bk, Aws, cpr);
  score_pool_kernel<<<dim3(TILES, B_), 256, SMEM, stream>>>(Kg, mask, Aws, cpr,
                                                            out_attn, Tpart);
  treduce_kernel<<<(B_*H_*D_)/256, 256, 0, stream>>>(Tpart, Tws);
  softmax_norm_kernel<<<B_*H_, 256, 0, stream>>>(out_attn, Ssum);
  final_kernel<<<B_, 256, 0, stream>>>(Tws, Ssum, Wv, bv, Wo, bo, out_pooled);
}
